// SAGE_4526895530523
// MI455X (gfx1250) — compile-verified, matrix operand order checked by reading, not run on hardware
//
#include <hip/hip_runtime.h>
#include <hip/hip_bf16.h>
#include <stdint.h>

// ---------------------------------------------------------------------------
// GraphSAGE (pool aggregator) forward for MI455X / gfx1250.
// GEMMs via v_wmma_f32_16x16x32_bf16 (wave32), segment-max via u32 atomics
// (valid because relu output >= 0 -> float bits monotonic as uint32).
//
// Round-2 refinements:
//  - Weights converted *transposed* (Wt[NC][128] bf16) so B fragments load
//    with the same two-b128 pattern as A fragments (no scalar u16 walks).
//  - All A fragments of a tile are loaded before the WMMA chain so the
//    compiler can clause the loads and use staged s_wait_loadcnt.
// ---------------------------------------------------------------------------

#define NN 60000          // nodes
#define FD 128            // feature dim (K for every GEMM)
#define NTILES (NN / 16)  // 3750 row tiles, exact

typedef __attribute__((ext_vector_type(16))) __bf16 v16bf;
typedef __attribute__((ext_vector_type(8)))  __bf16 v8bf;
typedef __attribute__((ext_vector_type(8)))  float  v8f;

static __device__ __forceinline__ __bf16 f2bf(float f) {
  unsigned u = __float_as_uint(f);
  unsigned r = (u + 0x7FFFu + ((u >> 16) & 1u)) >> 16;  // round-to-nearest-even
  unsigned short us = (unsigned short)r;
  return __builtin_bit_cast(__bf16, us);
}

// ------------------------- elementwise helpers -----------------------------

__global__ void f32_to_bf16_k(const float* __restrict__ in,
                              __bf16* __restrict__ out, long n) {
  long i = (long)blockIdx.x * blockDim.x + threadIdx.x;
  if (i < n) out[i] = f2bf(in[i]);
}

// W [128][NC] f32 (row-major)  ->  Wt [NC][128] bf16 (transposed)
__global__ void wT_bf16_k(const float* __restrict__ W,
                          __bf16* __restrict__ Wt, int NC) {
  int i = blockIdx.x * blockDim.x + threadIdx.x;
  if (i < FD * NC) {
    int k = i / NC, n = i % NC;
    Wt[n * FD + k] = f2bf(W[i]);
  }
}

__global__ void fill_zero_k(unsigned* __restrict__ p, long n) {
  long i = (long)blockIdx.x * blockDim.x + threadIdx.x;
  if (i < n) p[i] = 0u;
}

// ------------------------- edge scatter-max --------------------------------
// 32 threads per edge (one wave per edge, 512B coalesced row gather),
// 4 floats per thread. hp >= 0 so uint-max == float-max.
__global__ void edge_max_k(const float* __restrict__ hp,
                           const int* __restrict__ src,
                           const int* __restrict__ dst,
                           unsigned* __restrict__ agg, long nwork) {
  long i = (long)blockIdx.x * blockDim.x + threadIdx.x;
  if (i >= nwork) return;
  long e  = i >> 5;
  int  f4 = (int)(i & 31) * 4;
  const float4 v = *(const float4*)(hp + (long)src[e] * FD + f4);
  unsigned* a = agg + (long)dst[e] * FD + f4;
  atomicMax(a + 0, __float_as_uint(v.x));
  atomicMax(a + 1, __float_as_uint(v.y));
  atomicMax(a + 2, __float_as_uint(v.z));
  atomicMax(a + 3, __float_as_uint(v.w));
}

// ------------------------- WMMA fragment loader ----------------------------
// 16-bit 16x32 fragment layout (A and B identical once B comes from Wt):
// lanes 0-15 hold row (lane&15) with K {kbase+0..7, kbase+16..23},
// lanes 16-31 hold row (lane&15) with K {kbase+8..15, kbase+24..31};
// ext-vector element pairs (2i,2i+1) pack into VGPR i -> two b128 loads.
static __device__ __forceinline__ v16bf load_frag(const __bf16* __restrict__ P,
                                                  long row0, int kbase, int lane) {
  long r   = row0 + (lane & 15);
  int koff = kbase + ((lane & 16) ? 8 : 0);
  const __bf16* p = P + r * FD + koff;
  v8bf lo = *(const v8bf*)(p);        // K = koff..koff+7   (global_load_b128)
  v8bf hi = *(const v8bf*)(p + 16);   // K = koff+16..+23   (global_load_b128)
  v16bf f;
#pragma unroll
  for (int e = 0; e < 8; ++e) { f[e] = lo[e]; f[8 + e] = hi[e]; }
  return f;
}

// ------------------------- WMMA GEMM ---------------------------------------
// Block = (NC/16) waves; wave w owns output columns [w*16, w*16+16).
// B fragments (transposed weights) load once and stay in registers; the
// block grid-strides over 16-row tiles of A. DUAL fuses a second A@W product
// into the same accumulator (fc_self + fc_neigh).
template <int NC, bool RELU, bool DUAL>
__global__ void gemm_k(const __bf16* __restrict__ A1, const __bf16* __restrict__ W1t,
                       const __bf16* __restrict__ A2, const __bf16* __restrict__ W2t,
                       const float* __restrict__ bias, float* __restrict__ out,
                       int ntiles) {
  const int lane = threadIdx.x & 31;
  const int wave = threadIdx.x >> 5;
  const int col0 = wave * 16;

  v16bf b1[4], b2[4];
#pragma unroll
  for (int k = 0; k < 4; ++k) b1[k] = load_frag(W1t, col0, k * 32, lane);
  if (DUAL) {
#pragma unroll
    for (int k = 0; k < 4; ++k) b2[k] = load_frag(W2t, col0, k * 32, lane);
  }

  const int   n  = col0 + (lane & 15);
  const float bv = bias[n];

  for (int tile = blockIdx.x; tile < ntiles; tile += gridDim.x) {
    long row0 = (long)tile * 16;

    // Hoist all A loads so they clause together; waits then stage down.
    v16bf a1[4];
#pragma unroll
    for (int k = 0; k < 4; ++k) a1[k] = load_frag(A1, row0, k * 32, lane);
    v16bf a2[4];
    if (DUAL) {
#pragma unroll
      for (int k = 0; k < 4; ++k) a2[k] = load_frag(A2, row0, k * 32, lane);
    }

    v8f acc = {};
#pragma unroll
    for (int k = 0; k < 4; ++k)
      acc = __builtin_amdgcn_wmma_f32_16x16x32_bf16(false, a1[k], false, b1[k],
                                                    (short)0, acc, false, false);
    if (DUAL) {
#pragma unroll
      for (int k = 0; k < 4; ++k)
        acc = __builtin_amdgcn_wmma_f32_16x16x32_bf16(false, a2[k], false, b2[k],
                                                      (short)0, acc, false, false);
    }

    // C/D layout: lane<16 -> rows row0+0..7, lane>=16 -> rows row0+8..15, col n.
    long mbase = row0 + ((lane & 16) ? 8 : 0);
#pragma unroll
    for (int r = 0; r < 8; ++r) {
      float v = acc[r] + bv;
      if (RELU) v = fmaxf(v, 0.0f);
      out[(mbase + r) * NC + n] = v;
    }
  }
}

// ------------------------- tanh + BatchNorm --------------------------------
// Thread = feature column (coalesced). tanh applied in place, per-feature
// sum/sumsq accumulated via f32 atomics into sums[0..F) / sums[F..2F).
__global__ void bn_stats_k(float* __restrict__ h, float* __restrict__ sums,
                           long nrows) {
  int c = threadIdx.x;  // blockDim.x == FD
  float s = 0.0f, ss = 0.0f;
  for (long r = blockIdx.x; r < nrows; r += gridDim.x) {
    float v = tanhf(h[r * FD + c]);
    h[r * FD + c] = v;
    s += v;
    ss += v * v;
  }
  atomicAdd(&sums[c], s);
  atomicAdd(&sums[FD + c], ss);
}

__global__ void bn_finalize_k(const float* __restrict__ sums,
                              const float* __restrict__ g,
                              const float* __restrict__ be,
                              float* __restrict__ scsh, float invN) {
  int c = threadIdx.x;
  if (c < FD) {
    float mu  = sums[c] * invN;
    float var = sums[FD + c] * invN - mu * mu;
    float sc  = g[c] * rsqrtf(var + 1e-5f);
    scsh[c]      = sc;
    scsh[FD + c] = be[c] - mu * sc;
  }
}

__global__ void bn_apply_k(const float* __restrict__ h,
                           const float* __restrict__ scsh,
                           __bf16* __restrict__ out, long n) {
  long i = (long)blockIdx.x * blockDim.x + threadIdx.x;
  if (i < n) {
    int c = (int)(i & (FD - 1));
    out[i] = f2bf(h[i] * scsh[c] + scsh[FD + c]);
  }
}

// ------------------------- host driver -------------------------------------

static void run_layer(const __bf16* hin, const __bf16* Wpt, const float* bp,
                      const __bf16* Wst, const float* bs, const __bf16* Wnt,
                      int NC, float* outf,
                      float* hp, unsigned* agg, __bf16* aggb,
                      const int* src, const int* dst, long E, hipStream_t s) {
  // 1) hp = relu(hin @ Wp + bp)   [NN, 128]
  gemm_k<FD, true, false><<<NTILES, 256, 0, s>>>(hin, Wpt, nullptr, nullptr,
                                                 bp, hp, NTILES);
  // 2) agg = 0 ; segment-max over edges
  long na = (long)NN * FD;
  fill_zero_k<<<(int)((na + 255) / 256), 256, 0, s>>>(agg, na);
  long nwork = E * 32;
  edge_max_k<<<(int)((nwork + 255) / 256), 256, 0, s>>>(hp, src, dst, agg, nwork);
  // 3) agg -> bf16
  f32_to_bf16_k<<<(int)((na + 255) / 256), 256, 0, s>>>((const float*)agg, aggb, na);
  // 4) out = hin @ Ws + bs + agg @ Wn
  if (NC == FD) {
    gemm_k<FD, false, true><<<NTILES, 256, 0, s>>>(hin, Wst, aggb, Wnt, bs, outf, NTILES);
  } else {  // NC == 64
    gemm_k<64, false, true><<<NTILES, 128, 0, s>>>(hin, Wst, aggb, Wnt, bs, outf, NTILES);
  }
}

static void run_bn(float* outf, const float* g, const float* be,
                   float* stats, float* scsh, __bf16* hout, hipStream_t s) {
  fill_zero_k<<<1, 256, 0, s>>>((unsigned*)stats, 2 * FD);
  bn_stats_k<<<512, FD, 0, s>>>(outf, stats, NN);
  bn_finalize_k<<<1, FD, 0, s>>>(stats, g, be, scsh, 1.0f / (float)NN);
  long n = (long)NN * FD;
  bn_apply_k<<<(int)((n + 255) / 256), 256, 0, s>>>(outf, scsh, hout, n);
}

extern "C" void kernel_launch(void* const* d_in, const int* in_sizes, int n_in,
                              void* d_out, int out_size, void* d_ws, size_t ws_size,
                              hipStream_t stream) {
  const float* x   = (const float*)d_in[0];
  const int*   src = (const int*)d_in[1];
  const int*   dst = (const int*)d_in[2];
  const long   E   = in_sizes[1];

  const float* W1p = (const float*)d_in[3];  const float* b1p = (const float*)d_in[4];
  const float* W1s = (const float*)d_in[5];  const float* b1s = (const float*)d_in[6];
  const float* W1n = (const float*)d_in[7];
  const float* g1  = (const float*)d_in[8];  const float* be1 = (const float*)d_in[9];
  const float* W2p = (const float*)d_in[10]; const float* b2p = (const float*)d_in[11];
  const float* W2s = (const float*)d_in[12]; const float* b2s = (const float*)d_in[13];
  const float* W2n = (const float*)d_in[14];
  const float* g2  = (const float*)d_in[15]; const float* be2 = (const float*)d_in[16];
  const float* W3p = (const float*)d_in[17]; const float* b3p = (const float*)d_in[18];
  const float* W3s = (const float*)d_in[19]; const float* b3s = (const float*)d_in[20];
  const float* W3n = (const float*)d_in[21];

  // ---- carve workspace ----
  char* w = (char*)d_ws;
  auto carve = [&](size_t bytes) -> void* {
    void* p = (void*)w;
    w += (bytes + 255) & ~(size_t)255;
    return p;
  };
  const long NF = (long)NN * FD;
  __bf16*   hbuf  = (__bf16*)carve(NF * sizeof(__bf16));   // bf16 activations
  __bf16*   aggb  = (__bf16*)carve(NF * sizeof(__bf16));   // bf16 aggregate
  float*    hp    = (float*)carve(NF * sizeof(float));     // relu pool output
  unsigned* agg   = (unsigned*)carve(NF * sizeof(float));  // max accum (f32 bits)
  float*    outf  = (float*)carve(NF * sizeof(float));     // layer output f32
  float*    stats = (float*)carve(2 * FD * sizeof(float));
  float*    scsh  = (float*)carve(2 * FD * sizeof(float));

  // bf16 *transposed* weight copies: Wt[NC][128]
  const float* wsrc[9] = {W1p, W1s, W1n, W2p, W2s, W2n, W3p, W3s, W3n};
  const int    wnc[9]  = {128, 128, 128, 128, 128, 128, 128, 64, 64};
  __bf16* wb[9];
  for (int i = 0; i < 9; ++i) {
    long nel = (long)FD * wnc[i];
    wb[i] = (__bf16*)carve(nel * sizeof(__bf16));
    wT_bf16_k<<<(int)((nel + 255) / 256), 256, 0, stream>>>(wsrc[i], wb[i], wnc[i]);
  }

  // x -> bf16 activation buffer
  f32_to_bf16_k<<<(int)((NF + 255) / 256), 256, 0, stream>>>(x, hbuf, NF);

  // conv1 + tanh + BN
  run_layer(hbuf, wb[0], b1p, wb[1], b1s, wb[2], FD, outf,
            hp, agg, aggb, src, dst, E, stream);
  run_bn(outf, g1, be1, stats, scsh, hbuf, stream);

  // conv2 twice (shared weights) + tanh + BN
  for (int rep = 0; rep < 2; ++rep) {
    run_layer(hbuf, wb[3], b2p, wb[4], b2s, wb[5], FD, outf,
              hp, agg, aggb, src, dst, E, stream);
    run_bn(outf, g2, be2, stats, scsh, hbuf, stream);
  }

  // conv3 -> d_out (f32, [NN, 64]), no activation/norm
  run_layer(hbuf, wb[6], b3p, wb[7], b3s, wb[8], 64, (float*)d_out,
            hp, agg, aggb, src, dst, E, stream);

  (void)n_in; (void)out_size; (void)ws_size; (void)in_sizes;
}
